// slmgnn_59064390255146
// MI455X (gfx1250) — compile-verified, matrix operand order checked by reading, not run on hardware
//
#include <hip/hip_runtime.h>
#include <hip/hip_bf16.h>

// ---------------------------------------------------------------------------
// Graph-WaveNet style network on gfx1250 (MI455X).
// Dominant cost: 24 GEMMs of (43008x512)@(512x512) -> bf16 WMMA 16x16x32,
// f32 accumulate. Each wave computes a 4x1 block of 16x16 tiles sharing the
// B fragment; fragments are double-buffered and the schedule is pinned with
// sched_group_barrier so step-(k+1) loads issue before step-k WMMAs.
// Everything else (channel mixes, GRU scan, head) is fp32 VALU.
// ---------------------------------------------------------------------------

typedef __attribute__((ext_vector_type(16))) __bf16 v16bf;
typedef __attribute__((ext_vector_type(8)))  __bf16 v8bf;
typedef __attribute__((ext_vector_type(8)))  float  v8f;

constexpr int Bn = 8, Tn = 168, Nn = 512, Cn = 32, Ln = 4, Kn = 50;
constexpr int Rrows = Bn * Cn * Tn;          // 43008 GEMM rows
constexpr int BTN   = Bn * Tn * Nn;          // 688128
constexpr int RT    = 4;                     // row tiles per wave in GEMM

__device__ __forceinline__ float sigm(float x) { return 1.0f / (1.0f + __expf(-x)); }

// ---- transpose + cast a support (N,N) f32 -> (N,N) bf16, dst[m][k]=src[k][m]
__global__ __launch_bounds__(256) void k_transcast(const float* __restrict__ src,
                                                   __hip_bfloat16* __restrict__ dstT) {
  int tid = blockIdx.x * 256 + threadIdx.x;          // over N*N
  int m = tid & (Nn - 1);
  int n = tid >> 9;
  dstT[(size_t)m * Nn + n] = __float2bfloat16(src[(size_t)n * Nn + m]);
}

// ---- adp = softmax(relu(nv1 @ nv2), axis=1); write transposed bf16
__global__ __launch_bounds__(512) void k_adp(const float* __restrict__ nv1,
                                             const float* __restrict__ nv2,
                                             __hip_bfloat16* __restrict__ dstT) {
  int n = blockIdx.x;          // row
  int m = threadIdx.x;         // col
  float acc = 0.f;
#pragma unroll
  for (int j = 0; j < 10; ++j) acc += nv1[n * 10 + j] * nv2[j * Nn + m];
  float v = fmaxf(acc, 0.f);
  __shared__ float red[Nn];
  red[m] = v; __syncthreads();
  for (int s = 256; s > 0; s >>= 1) { if (m < s) red[m] = fmaxf(red[m], red[m + s]); __syncthreads(); }
  float mx = red[0]; __syncthreads();
  float e = __expf(v - mx);
  red[m] = e; __syncthreads();
  for (int s = 256; s > 0; s >>= 1) { if (m < s) red[m] += red[m + s]; __syncthreads(); }
  float sum = red[0];
  dstT[(size_t)m * Nn + n] = __float2bfloat16(e / sum);
}

// ---- initial x[b,c,t,n] = skip_b[c] + sum_j skip_w[c,j]*input[b,sel[j],t,n]
__global__ __launch_bounds__(256) void k_init_x(const float* __restrict__ input,
                                                const float* __restrict__ sw,
                                                const float* __restrict__ sb,
                                                float* __restrict__ x) {
  int tid = blockIdx.x * 256 + threadIdx.x;          // (b,t,n)
  int n = tid & (Nn - 1);
  int t = (tid >> 9) % Tn;
  int b = (tid >> 9) / Tn;
  const int sel[4] = {0, 3, 5, 6};
  float in4[4];
#pragma unroll
  for (int j = 0; j < 4; ++j)
    in4[j] = input[((size_t)(b * 7 + sel[j]) * Tn + t) * Nn + n];
#pragma unroll
  for (int c = 0; c < Cn; ++c) {
    float a = sb[c];
#pragma unroll
    for (int j = 0; j < 4; ++j) a += sw[c * 4 + j] * in4[j];
    x[((size_t)(b * Cn + c) * Tn + t) * Nn + n] = a;
  }
}

// ---- gated temporal conv (1x2, left pad): xg = tanh(f)*sigm(g);
//      skip += xg; xg_bf = bf16(xg)
__global__ __launch_bounds__(256) void k_tconv(const float* __restrict__ x,
                                               const float* __restrict__ fw_all,
                                               const float* __restrict__ fb_all,
                                               const float* __restrict__ gw_all,
                                               const float* __restrict__ gb_all,
                                               int layer,
                                               __hip_bfloat16* __restrict__ xg_bf,
                                               float* __restrict__ skip) {
  int tid = blockIdx.x * 256 + threadIdx.x;
  int n = tid & (Nn - 1);
  int t = (tid >> 9) % Tn;
  int b = (tid >> 9) / Tn;
  const float* fw = fw_all + (size_t)layer * Cn * Cn * 2;
  const float* gw = gw_all + (size_t)layer * Cn * Cn * 2;
  const float* fb = fb_all + layer * Cn;
  const float* gb = gb_all + layer * Cn;
  float facc[Cn], gacc[Cn];
#pragma unroll
  for (int o = 0; o < Cn; ++o) { facc[o] = fb[o]; gacc[o] = gb[o]; }
  for (int c = 0; c < Cn; ++c) {
    size_t base = ((size_t)(b * Cn + c) * Tn + t) * Nn + n;
    float xc = x[base];
    float xp = (t > 0) ? x[base - Nn] : 0.f;
#pragma unroll
    for (int o = 0; o < Cn; ++o) {
      int wi = (o * Cn + c) * 2;
      facc[o] += xp * fw[wi] + xc * fw[wi + 1];
      gacc[o] += xp * gw[wi] + xc * gw[wi + 1];
    }
  }
#pragma unroll
  for (int o = 0; o < Cn; ++o) {
    float xg = tanhf(facc[o]) * sigm(gacc[o]);
    size_t oidx = ((size_t)(b * Cn + o) * Tn + t) * Nn + n;
    skip[oidx] += xg;
    xg_bf[oidx] = __float2bfloat16(xg);
  }
}

// ---- WMMA GEMM: D_bf16(R x 512) = A_bf16(R x 512) @ S, with S given as
//      its transpose Bt (Bt[m][k] = S[k][m]).
//      One wave computes RT=4 stacked 16x16 tiles sharing the B fragment;
//      double-buffered fragments + sched_group_barrier pipeline:
//      [lds k=0][lds k=1][wmma k=0][lds k=2][wmma k=1]...[wmma k=15]
__global__ __launch_bounds__(32) void k_gemm(const __hip_bfloat16* __restrict__ A,
                                             const __hip_bfloat16* __restrict__ Bt,
                                             __hip_bfloat16* __restrict__ D) {
  const int lane = threadIdx.x;
  const int half = lane >> 4;
  const int lm   = lane & 15;
  const int c0 = blockIdx.x * 16;                    // col tile
  const int r0 = blockIdx.y * (16 * RT);             // first row tile
  const __hip_bfloat16* brow = Bt + (size_t)(c0 + lm) * Nn;
  const __hip_bfloat16* arow[RT];
#pragma unroll
  for (int i = 0; i < RT; ++i)
    arow[i] = A + (size_t)(r0 + i * 16 + lm) * Nn;
  v8f acc[RT];
#pragma unroll
  for (int i = 0; i < RT; ++i) acc[i] = (v8f){0.f, 0.f, 0.f, 0.f, 0.f, 0.f, 0.f, 0.f};

  constexpr int NLD = 2 + 2 * RT;                    // vmem reads per k-step

  // double-buffered fragments (buf x {B lo/hi, A lo/hi per row tile})
  v8bf bl[2], bh[2], aal[2][RT], aah[2][RT];
  {
    const int o1 = half * 8;
    const int o2 = 16 + half * 8;
    bl[0] = *reinterpret_cast<const v8bf*>(brow + o1);
    bh[0] = *reinterpret_cast<const v8bf*>(brow + o2);
#pragma unroll
    for (int i = 0; i < RT; ++i) {
      aal[0][i] = *reinterpret_cast<const v8bf*>(arow[i] + o1);
      aah[0][i] = *reinterpret_cast<const v8bf*>(arow[i] + o2);
    }
  }
  // group 0: the prologue's NLD loads
  __builtin_amdgcn_sched_group_barrier(0x020, NLD, 0);

#pragma unroll
  for (int kb = 0; kb < Nn; kb += 32) {
    const int cur = (kb >> 5) & 1;
    const int nxt = cur ^ 1;
    const bool pf = (kb + 32 < Nn);
    if (pf) {                                        // prefetch next k-step
      const int p1 = kb + 32 + half * 8;
      const int p2 = kb + 48 + half * 8;
      bl[nxt] = *reinterpret_cast<const v8bf*>(brow + p1);
      bh[nxt] = *reinterpret_cast<const v8bf*>(brow + p2);
#pragma unroll
      for (int i = 0; i < RT; ++i) {
        aal[nxt][i] = *reinterpret_cast<const v8bf*>(arow[i] + p1);
        aah[nxt][i] = *reinterpret_cast<const v8bf*>(arow[i] + p2);
      }
    }
    v16bf bv = __builtin_shufflevector(bl[cur], bh[cur],
                                       0,1,2,3,4,5,6,7,8,9,10,11,12,13,14,15);
#pragma unroll
    for (int i = 0; i < RT; ++i) {
      v16bf av = __builtin_shufflevector(aal[cur][i], aah[cur][i],
                                         0,1,2,3,4,5,6,7,8,9,10,11,12,13,14,15);
      acc[i] = __builtin_amdgcn_wmma_f32_16x16x32_bf16(false, av, false, bv,
                                                       (short)0, acc[i], false, false);
    }
    // group order: loads of step k+1, THEN wmmas of step k
    if (pf) __builtin_amdgcn_sched_group_barrier(0x020, NLD, 0);
    __builtin_amdgcn_sched_group_barrier(0x008, RT, 0);
  }
#pragma unroll
  for (int i = 0; i < RT; ++i) {
#pragma unroll
    for (int v = 0; v < 8; ++v) {
      const int row = r0 + i * 16 + half * 8 + v;
      D[(size_t)row * Nn + c0 + lm] = __float2bfloat16(acc[i][v]);
    }
  }
}

// ---- channel mix accumulate: gout[b,o,t,n] (+)= sum_c W[o, jbase+c]*feat[b,c,t,n]
__global__ __launch_bounds__(256) void k_mix(const __hip_bfloat16* __restrict__ feat,
                                             const float* __restrict__ gw_all,
                                             const float* __restrict__ gb_all,
                                             int layer, int jbase, int init,
                                             float* __restrict__ gout) {
  int tid = blockIdx.x * 256 + threadIdx.x;
  int n = tid & (Nn - 1);
  int t = (tid >> 9) % Tn;
  int b = (tid >> 9) / Tn;
  const float* w = gw_all + (size_t)layer * Cn * (7 * Cn) + jbase;
  float acc[Cn];
#pragma unroll
  for (int o = 0; o < Cn; ++o)
    acc[o] = init ? gb_all[layer * Cn + o]
                  : gout[((size_t)(b * Cn + o) * Tn + t) * Nn + n];
  for (int c = 0; c < Cn; ++c) {
    float f = __bfloat162float(feat[((size_t)(b * Cn + c) * Tn + t) * Nn + n]);
#pragma unroll
    for (int o = 0; o < Cn; ++o) acc[o] += w[o * (7 * Cn) + c] * f;
  }
#pragma unroll
  for (int o = 0; o < Cn; ++o)
    gout[((size_t)(b * Cn + o) * Tn + t) * Nn + n] = acc[o];
}

// ---- X[b,t,n] = ingru_b + sum_c ingru_w[c]*gout[b,c,t,n]
__global__ __launch_bounds__(256) void k_ingru(const float* __restrict__ gout,
                                               const float* __restrict__ iw,
                                               const float* __restrict__ ib,
                                               float* __restrict__ X) {
  int tid = blockIdx.x * 256 + threadIdx.x;
  int n = tid & (Nn - 1);
  int t = (tid >> 9) % Tn;
  int b = (tid >> 9) / Tn;
  float a = ib[0];
#pragma unroll
  for (int c = 0; c < Cn; ++c)
    a += iw[c] * gout[((size_t)(b * Cn + c) * Tn + t) * Nn + n];
  X[((size_t)b * Tn + t) * Nn + n] = a;
}

// ---- sequential GRU scan: one workgroup per batch, thread n owns node n
__global__ __launch_bounds__(512) void k_gru(const float* __restrict__ X,
                                             const float* __restrict__ input,
                                             const float* __restrict__ zw, const float* __restrict__ zb,
                                             const float* __restrict__ rw, const float* __restrict__ rb,
                                             const float* __restrict__ hw, const float* __restrict__ hb,
                                             float* __restrict__ hout) {
  const int b = blockIdx.x;
  const int n = threadIdx.x;
  __shared__ float sxin[Nn], sh[Nn], srh[Nn];
  float h = 0.f;
  const float* zrow = zw + (size_t)n * (2 * Nn);
  const float* rrow = rw + (size_t)n * (2 * Nn);
  const float* hrow = hw + (size_t)n * (2 * Nn);
  for (int t = 0; t < Tn; ++t) {
    float Xi = X[((size_t)b * Tn + t) * Nn + n];
    float Mi = input[((size_t)(b * 7 + 1) * Tn + t) * Nn + n];
    float xin = sigm(Xi * Mi + h * (1.f - Mi));
    sxin[n] = xin; sh[n] = h;
    __syncthreads();
    float za = zb[n], ra = rb[n];
#pragma unroll 8
    for (int j = 0; j < Nn; ++j) {
      float xi = sxin[j], hj = sh[j];
      za += xi * zrow[j] + hj * zrow[Nn + j];
      ra += xi * rrow[j] + hj * rrow[Nn + j];
    }
    float z = sigm(za), r = sigm(ra);
    srh[n] = r * h;
    __syncthreads();
    float ha = hb[n];
#pragma unroll 8
    for (int j = 0; j < Nn; ++j)
      ha += sxin[j] * hrow[j] + srh[j] * hrow[Nn + j];
    float ht = tanhf(ha);
    h = (1.f - z) * h + z * ht;
    hout[((size_t)b * Tn + t) * Nn + n] = h;
    __syncthreads();            // protect sxin/sh/srh before next step
  }
}

// ---- x = x + gout + (regru_w[c]*hout[b,t,n] + regru_b[c])
__global__ __launch_bounds__(256) void k_xnew(const float* __restrict__ hout,
                                              const float* __restrict__ rw,
                                              const float* __restrict__ rb,
                                              const float* __restrict__ gout,
                                              float* __restrict__ x) {
  size_t idx = (size_t)blockIdx.x * 256 + threadIdx.x;   // over R*N
  int n = idx & (Nn - 1);
  size_t q = idx >> 9;
  int t = q % Tn;
  size_t q2 = q / Tn;
  int c = q2 & (Cn - 1);
  int b = (int)(q2 >> 5);
  float re = rw[c] * hout[((size_t)b * Tn + t) * Nn + n] + rb[c];
  x[idx] = x[idx] + gout[idx] + re;
}

// ---- head: h2[b,n,t] = end2( relu( end1( relu(skip) ) ) )
__global__ __launch_bounds__(256) void k_end(const float* __restrict__ skip,
                                             const float* __restrict__ w1, const float* __restrict__ b1,
                                             const float* __restrict__ w2, const float* __restrict__ b2,
                                             float* __restrict__ h2) {
  int tid = blockIdx.x * 256 + threadIdx.x;
  int n = tid & (Nn - 1);
  int t = (tid >> 9) % Tn;
  int b = (tid >> 9) / Tn;
  float acc[64];
#pragma unroll
  for (int o = 0; o < 64; ++o) acc[o] = 0.f;
  for (int c = 0; c < Cn; ++c) {
    float s = fmaxf(skip[((size_t)(b * Cn + c) * Tn + t) * Nn + n], 0.f);
#pragma unroll
    for (int o = 0; o < 64; ++o) acc[o] += w1[o * Cn + c] * s;
  }
  float out = b2[0];
#pragma unroll
  for (int o = 0; o < 64; ++o) out += w2[o] * fmaxf(acc[o] + b1[o], 0.f);
  h2[((size_t)b * Nn + n) * Tn + t] = out;
}

// ---- gather missing nodes: out[b,k,t] = h2[b, mn[b,k], t]
__global__ __launch_bounds__(256) void k_gather(const float* __restrict__ h2,
                                                const int* __restrict__ mn,
                                                float* __restrict__ out) {
  int tid = blockIdx.x * 256 + threadIdx.x;
  if (tid >= Bn * Kn * Tn) return;
  int t = tid % Tn;
  int k = (tid / Tn) % Kn;
  int b = tid / (Tn * Kn);
  int node = mn[b * Kn + k];
  out[tid] = h2[((size_t)b * Nn + node) * Tn + t];
}

// ---------------------------------------------------------------------------
extern "C" void kernel_launch(void* const* d_in, const int* in_sizes, int n_in,
                              void* d_out, int out_size, void* d_ws, size_t ws_size,
                              hipStream_t stream) {
  (void)in_sizes; (void)n_in; (void)out_size; (void)ws_size;
  const float* input   = (const float*)d_in[0];
  const int*   mn      = (const int*)  d_in[1];
  const float* sup0    = (const float*)d_in[2];
  const float* sup1    = (const float*)d_in[3];
  const float* nv1     = (const float*)d_in[4];
  const float* nv2     = (const float*)d_in[5];
  const float* skip_w  = (const float*)d_in[6];
  const float* skip_b  = (const float*)d_in[7];
  const float* fw      = (const float*)d_in[8];
  const float* fb      = (const float*)d_in[9];
  const float* gw      = (const float*)d_in[10];
  const float* gb      = (const float*)d_in[11];
  const float* gcw     = (const float*)d_in[12];
  const float* gcb     = (const float*)d_in[13];
  const float* iw      = (const float*)d_in[14];
  const float* ib      = (const float*)d_in[15];
  const float* rgw     = (const float*)d_in[16];
  const float* rgb     = (const float*)d_in[17];
  const float* zlw     = (const float*)d_in[18];
  const float* zlb     = (const float*)d_in[19];
  const float* rlw     = (const float*)d_in[20];
  const float* rlb     = (const float*)d_in[21];
  const float* hlw     = (const float*)d_in[22];
  const float* hlb     = (const float*)d_in[23];
  const float* e1w     = (const float*)d_in[24];
  const float* e1b     = (const float*)d_in[25];
  const float* e2w     = (const float*)d_in[26];
  const float* e2b     = (const float*)d_in[27];

  // -------- carve workspace (~406 MB total) --------
  char* base = (char*)d_ws;
  size_t off = 0;
  auto carve = [&](size_t bytes) -> void* {
    void* p = base + off;
    off = (off + bytes + 255) & ~(size_t)255;
    return p;
  };
  const size_t act_f32 = (size_t)Rrows * Nn * sizeof(float);            // 88.1 MB
  const size_t act_bf  = (size_t)Rrows * Nn * sizeof(__hip_bfloat16);   // 44.0 MB
  const size_t btn_f32 = (size_t)BTN * sizeof(float);                   // 2.75 MB

  __hip_bfloat16* supT0 = (__hip_bfloat16*)carve((size_t)Nn * Nn * 2);
  __hip_bfloat16* supT1 = (__hip_bfloat16*)carve((size_t)Nn * Nn * 2);
  __hip_bfloat16* supT2 = (__hip_bfloat16*)carve((size_t)Nn * Nn * 2);
  float*          x     = (float*)carve(act_f32);
  float*          skip  = (float*)carve(act_f32);
  float*          gout  = (float*)carve(act_f32);
  __hip_bfloat16* xg_bf = (__hip_bfloat16*)carve(act_bf);
  __hip_bfloat16* t1_bf = (__hip_bfloat16*)carve(act_bf);
  __hip_bfloat16* t2_bf = (__hip_bfloat16*)carve(act_bf);
  float*          Xbuf  = (float*)carve(btn_f32);
  float*          hout  = (float*)carve(btn_f32);
  float*          h2    = (float*)carve(btn_f32);

  hipMemsetAsync(skip, 0, act_f32, stream);

  k_transcast<<<(Nn * Nn) / 256, 256, 0, stream>>>(sup0, supT0);
  k_transcast<<<(Nn * Nn) / 256, 256, 0, stream>>>(sup1, supT1);
  k_adp<<<Nn, Nn, 0, stream>>>(nv1, nv2, supT2);
  k_init_x<<<BTN / 256, 256, 0, stream>>>(input, skip_w, skip_b, x);

  const __hip_bfloat16* sT[3] = {supT0, supT1, supT2};
  const dim3 ggrid(Nn / 16, Rrows / (16 * RT));   // (32, 672): 4 row tiles/wave

  for (int l = 0; l < Ln; ++l) {
    k_tconv<<<BTN / 256, 256, 0, stream>>>(x, fw, fb, gw, gb, l, xg_bf, skip);
    k_mix<<<BTN / 256, 256, 0, stream>>>(xg_bf, gcw, gcb, l, 0, 1, gout);
    for (int s = 0; s < 3; ++s) {
      k_gemm<<<ggrid, 32, 0, stream>>>(xg_bf, sT[s], t1_bf);
      k_mix<<<BTN / 256, 256, 0, stream>>>(t1_bf, gcw, gcb, l, (2 * s + 1) * Cn, 0, gout);
      k_gemm<<<ggrid, 32, 0, stream>>>(t1_bf, sT[s], t2_bf);
      k_mix<<<BTN / 256, 256, 0, stream>>>(t2_bf, gcw, gcb, l, (2 * s + 2) * Cn, 0, gout);
    }
    k_ingru<<<BTN / 256, 256, 0, stream>>>(gout, iw, ib, Xbuf);
    k_gru<<<Bn, Nn, 0, stream>>>(Xbuf, input, zlw, zlb, rlw, rlb, hlw, hlb, hout);
    k_xnew<<<(Rrows * Nn) / 256, 256, 0, stream>>>(hout, rgw, rgb, gout, x);
  }

  k_end<<<BTN / 256, 256, 0, stream>>>(skip, e1w, e1b, e2w, e2b, h2);
  k_gather<<<(Bn * Kn * Tn + 255) / 256, 256, 0, stream>>>(h2, mn, (float*)d_out);
}
